// VisionPooler_54065048322465
// MI455X (gfx1250) — matrix operand order repaired and checked by reading, not compiled, not run on hardware
//
#include <hip/hip_runtime.h>
#include <hip/hip_bf16.h>

typedef __attribute__((ext_vector_type(16))) _Float16 v16h;
typedef __attribute__((ext_vector_type(8)))  _Float16 v8h;
typedef __attribute__((ext_vector_type(8)))  float    v8f;
typedef __attribute__((ext_vector_type(2)))  _Float16 h2;

#define B_  16
#define L_  4096
#define D_  1152
#define O_  256
#define K_  4

#define DCH   64   // d columns per workgroup -> 18*16 = 288 WGs
#define NT    (DCH / 16)          // d-tiles per wave = 4
#define KT    32                  // K-chunk (16x16x32 f16 WMMA)
#define LSH   40                  // LDS row stride in halves (80B = 16B-aligned, odd*5 banks)
#define UNITS ((KT * DCH / 2) / 512)  // packed f16x2 units per thread = 2

// ---- kernel 0: zero valid flags (ws is poisoned, must self-init) ----
__global__ void vp_zero_valid(int* __restrict__ valid) {
    int t = blockIdx.x * blockDim.x + threadIdx.x;
    if (t < B_ * O_) valid[t] = 0;
}

// ---- kernel 1: per-batch max over clamped x position ----
__global__ void vp_maxx(const int* __restrict__ pos, int* __restrict__ maxx) {
    __shared__ int red[256];
    int b = blockIdx.x;
    int t = threadIdx.x;
    int m = 0;
    for (int l = t; l < L_; l += 256) {
        int x = pos[((size_t)b * L_ + l) * 2 + 0];
        if (x < 0) x = 0;
        if (x > m) m = x;
    }
    red[t] = m;
    __syncthreads();
    for (int s = 128; s > 0; s >>= 1) {
        if (t < s) red[t] = max(red[t], red[t + s]);
        __syncthreads();
    }
    if (t == 0) maxx[b] = red[0];
}

// ---- kernel 2: bucket index per (b,l); valid flags from raw index;
//      effective index folds padding (pad -> -1 -> zero one-hot weight) ----
__global__ void vp_index(const int* __restrict__ pos,
                         const unsigned char* __restrict__ pad,
                         const int* __restrict__ maxx,
                         int* __restrict__ idx, int* __restrict__ valid) {
    int t = blockIdx.x * blockDim.x + threadIdx.x;
    if (t >= B_ * L_) return;
    int b  = t / L_;
    int cx = max(pos[(size_t)t * 2 + 0], 0);
    int cy = max(pos[(size_t)t * 2 + 1], 0);
    int mx4 = (maxx[b] + 1) / K_;
    int id  = cx / K_ + mx4 * (cy / K_);
    if (id >= 0 && id < O_) valid[b * O_ + id] = 1;   // benign races, all write 1
    idx[t] = (id >= 0 && id < O_ && !pad[t]) ? id : -1;
}

// ---- kernel 3: one-hot pooling as dense f16 WMMA, hs read from HBM once,
//      2-deep LDS ping-pong pipeline, b128 B-fragment loads ----
__global__ __launch_bounds__(512)
void vp_pool(const float* __restrict__ hs, const int* __restrict__ idx,
             float* __restrict__ out) {
    __shared__ _Float16 lsB[2][DCH * LSH];  // transposed tile: [d][k], f16
    __shared__ int lsIdx[2][KT];

    const int b     = blockIdx.y;
    const int dbase = blockIdx.x * DCH;
    const int tid   = threadIdx.x;
    const int wave  = tid >> 5;
    const int lane  = tid & 31;
    const int obase = wave << 4;      // 16 waves cover all 256 output buckets

    const int h = lane >> 4;          // lane half
    const int m = lane & 15;          // A: row M / B,C: column N

    v8f c[NT] = {};

    const size_t hsBase = ((size_t)b * L_) * D_ + dbase;
    const int idxBase = b * L_;

    float r0[UNITS], r1[UNITS];
    int ridx = -1;

    // ---- prologue: load + stage chunk 0 ----
    #pragma unroll
    for (int i = 0; i < UNITS; ++i) {
        int u = i * 512 + tid;
        int kp = u >> 6;              // k-pair 0..15
        int d  = u & (DCH - 1);
        size_t g = hsBase + (size_t)(2 * kp) * D_ + d;
        r0[i] = hs[g];
        r1[i] = hs[g + D_];
    }
    if (tid < KT) ridx = idx[idxBase + tid];
    #pragma unroll
    for (int i = 0; i < UNITS; ++i) {
        int u = i * 512 + tid;
        int kp = u >> 6;
        int d  = u & (DCH - 1);
        h2 pk; pk.x = (_Float16)r0[i]; pk.y = (_Float16)r1[i];
        *(h2*)&lsB[0][d * LSH + 2 * kp] = pk;
    }
    if (tid < KT) lsIdx[0][tid] = ridx;
    __syncthreads();

    for (int kc = 0; kc < L_ / KT; ++kc) {
        const int cur = kc & 1;
        const int nxt = cur ^ 1;
        const bool more = (kc + 1) < (L_ / KT);

        // ---- issue global loads for next chunk (latency hidden by WMMA) ----
        if (more) {
            const int l0n = (kc + 1) * KT;
            #pragma unroll
            for (int i = 0; i < UNITS; ++i) {
                int u = i * 512 + tid;
                int kp = u >> 6;
                int d  = u & (DCH - 1);
                size_t g = hsBase + (size_t)(l0n + 2 * kp) * D_ + d;
                r0[i] = hs[g];
                r1[i] = hs[g + D_];
            }
            if (tid < KT) ridx = idx[idxBase + l0n + tid];
        }

        // ---- build one-hot A tile in the 16-bit A VGPR layout:
        //      lane m = row M; VGPR v half p <-> K = 2(v&3) + 16(v>>2) + 8h + p
        v16h a;
        #pragma unroll
        for (int v = 0; v < 8; ++v) {
            #pragma unroll
            for (int p = 0; p < 2; ++p) {
                int kk = ((v & 3) << 1) + ((v >> 2) << 4) + (h << 3) + p;
                a[2 * v + p] = (lsIdx[cur][kk] == obase + m) ? (_Float16)1.0f
                                                             : (_Float16)0.0f;
            }
        }

        // ---- preload ALL B tiles (distinct regs -> no WMMA WAR stalls),
        //      each fragment = two 8-half runs (K = 8h.., 16+8h..) = two ds_load_b128
        v16h bm[NT];
        #pragma unroll
        for (int t = 0; t < NT; ++t) {
            int d = t * 16 + m;       // column N supplied by this lane
            const v8h* bp = (const v8h*)&lsB[cur][d * LSH + (h << 3)];
            v8h lo = bp[0];
            v8h hi = bp[2];
            #pragma unroll
            for (int j = 0; j < 8; ++j) { bm[t][j] = lo[j]; bm[t][8 + j] = hi[j]; }
        }

        // ---- back-to-back WMMAs (independent accumulators, no hazards) ----
        #pragma unroll
        for (int t = 0; t < NT; ++t) {
            c[t] = __builtin_amdgcn_wmma_f32_16x16x32_f16(
                false, a, false, bm[t], (short)0, c[t], false, false);
        }

        // ---- stage next chunk into the other LDS buffer ----
        if (more) {
            #pragma unroll
            for (int i = 0; i < UNITS; ++i) {
                int u = i * 512 + tid;
                int kp = u >> 6;
                int d  = u & (DCH - 1);
                h2 pk; pk.x = (_Float16)r0[i]; pk.y = (_Float16)r1[i];
                *(h2*)&lsB[nxt][d * LSH + 2 * kp] = pk;
            }
            if (tid < KT) lsIdx[nxt][tid] = ridx;
        }
        __syncthreads();
    }

    // ---- store pooled * sqrt(D)/k^2. C layout: VGPR r -> M = r + 8h, N = m
    const float scale = 2.12132034355964f;  // sqrt(1152)/16
    #pragma unroll
    for (int t = 0; t < NT; ++t) {
        #pragma unroll
        for (int r = 0; r < 8; ++r) {
            int mm = r + (h << 3);
            out[((size_t)(b * O_) + obase + mm) * D_ + dbase + t * 16 + m] =
                c[t][r] * scale;
        }
    }
}

// ---- kernel 4: valid mask -> float, appended after pooled block ----
__global__ void vp_valid_out(const int* __restrict__ valid, float* __restrict__ out) {
    int t = blockIdx.x * blockDim.x + threadIdx.x;
    if (t < B_ * O_) out[t] = valid[t] ? 1.0f : 0.0f;
}

extern "C" void kernel_launch(void* const* d_in, const int* in_sizes, int n_in,
                              void* d_out, int out_size, void* d_ws, size_t ws_size,
                              hipStream_t stream) {
    const float*         hs  = (const float*)d_in[0];
    const int*           pos = (const int*)d_in[1];
    const unsigned char* pad = (const unsigned char*)d_in[2];
    // d_in[3] = output_length (256), baked into constants

    int* ws_maxx  = (int*)d_ws;            // 16 ints (padded to 64)
    int* ws_valid = ws_maxx + 64;          // B*O ints
    int* ws_idx   = ws_valid + B_ * O_;    // B*L ints

    float* out_pool  = (float*)d_out;
    float* out_valid = out_pool + (size_t)B_ * O_ * D_;

    vp_zero_valid<<<(B_ * O_ + 255) / 256, 256, 0, stream>>>(ws_valid);
    vp_maxx<<<B_, 256, 0, stream>>>(pos, ws_maxx);
    vp_index<<<(B_ * L_ + 255) / 256, 256, 0, stream>>>(pos, pad, ws_maxx,
                                                        ws_idx, ws_valid);
    dim3 grid(D_ / DCH, B_);
    vp_pool<<<grid, 512, 0, stream>>>(hs, ws_idx, out_pool);
    vp_valid_out<<<(B_ * O_ + 255) / 256, 256, 0, stream>>>(ws_valid, out_valid);
}
